// DSVT_78340203479623
// MI455X (gfx1250) — compile-verified, matrix operand order checked by reading, not run on hardware
//
#include <hip/hip_runtime.h>

// ---------------------------------------------------------------------------
// DSVT forward for MI455X (gfx1250): bf16 WMMA GEMMs (LDS-staged weights) +
// VALU attention + layernorms.
//   - weights pre-converted to bf16 once per launch
//   - 2D-grid GEMM: 8 waves/block share a 64-col weight tile staged in LDS
//     (padded stride -> conflict-optimal ds_load_b128, immediate offsets)
//   - K templated (192/384) -> fully unrolled hot loop
//   - GEMM-only intermediates (attn-out, FFN hidden) stored bf16
// ---------------------------------------------------------------------------

constexpr int kC    = 192;
constexpr int kDH   = 24;      // kC / heads(8)
constexpr int kFF   = 384;
constexpr int kSET  = 36;
constexpr int kNSET = 2048;
constexpr int kN    = kNSET * kSET;  // 73728
constexpr int kNL   = 8;             // layers
constexpr float kEPS   = 1e-5f;
constexpr float kSCALE = 0.2041241452319315f;  // 1/sqrt(24)

typedef __attribute__((ext_vector_type(16))) __bf16 v16bf;
typedef __attribute__((ext_vector_type(8)))  __bf16 v8bf;
typedef __attribute__((ext_vector_type(8)))  float  v8f;

// ---------------------------------------------------------------------------
// Fragment loaders (CDNA5 ISA 7.12.2, wave32):
//   A/B 16x32 bf16 frag for a lane = K values {kk+hi8 .. +7} and
//   {kk+16+hi8 .. +7}  (hi8 = 8 for lanes 16..31)  -> two contiguous 16B runs.
// ---------------------------------------------------------------------------
__device__ __forceinline__ v16bf frag_f32(const float* __restrict__ base,
                                          int kk, int hi8)
{
    v16bf f;
    #pragma unroll
    for (int j = 0; j < 8; ++j) {
        const int kOff = 2 * j + ((j & 4) ? 8 : 0) + hi8;
        const float2 p = *(const float2*)(base + kk + kOff);
        f[2 * j]     = (__bf16)p.x;
        f[2 * j + 1] = (__bf16)p.y;
    }
    return f;
}

__device__ __forceinline__ v16bf frag_f32_add(const float* __restrict__ b0,
                                              const float* __restrict__ b1,
                                              int kk, int hi8)
{
    v16bf f;
    #pragma unroll
    for (int j = 0; j < 8; ++j) {
        const int kOff = 2 * j + ((j & 4) ? 8 : 0) + hi8;
        float2 p = *(const float2*)(b0 + kk + kOff);
        const float2 q = *(const float2*)(b1 + kk + kOff);
        p.x += q.x; p.y += q.y;
        f[2 * j]     = (__bf16)p.x;
        f[2 * j + 1] = (__bf16)p.y;
    }
    return f;
}

__device__ __forceinline__ v16bf frag_bf16(const __bf16* __restrict__ base,
                                           int kk, int hi8)
{
    const v8bf c0 = *(const v8bf*)(base + kk + hi8);
    const v8bf c1 = *(const v8bf*)(base + kk + 16 + hi8);
    v16bf f;
    #pragma unroll
    for (int j = 0; j < 8; ++j) { f[j] = c0[j]; f[8 + j] = c1[j]; }
    return f;
}

// ---------------------------------------------------------------------------
// WMMA GEMM:  Y[M,Nout] = (X (+X2)) [M,K] @ W[Nout,K]^T + bias  (opt ReLU)
// Grid: (M/128, Nout/64); block = 8 waves; wave w -> m-tile blockIdx.x*8+w,
// all waves share the 64-col W tile staged in LDS (bf16, padded row stride).
// ---------------------------------------------------------------------------
template <bool XBF, bool OUTBF, bool ADDX2, int K>
__global__ __launch_bounds__(256)
void dsvt_gemm_wmma(const void* __restrict__ Xp, const float* __restrict__ X2,
                    const __bf16* __restrict__ W, const float* __restrict__ Bias,
                    void* __restrict__ Yp, int Nout, int relu)
{
    constexpr int KP = K + 8;              // padded LDS row stride (bf16)
    __shared__ __bf16 wtile[64 * KP];

    const int tid  = threadIdx.x;
    const int lane = tid & 31;
    const int wv   = tid >> 5;
    const int l15  = lane & 15;
    const int hi8  = ((lane >> 4) & 1) * 8;
    const int mt   = blockIdx.x * 8 + wv;
    const int n0   = blockIdx.y * 64;
    const int row  = mt * 16 + l15;

    // cooperative stage of W[n0 .. n0+63][0..K) into LDS (16B chunks)
    constexpr int CH = K / 8;              // chunks per row
    for (int i = tid; i < 64 * CH; i += 256) {
        const int r = i / CH;
        const int c = (i - r * CH) * 8;
        *(uint4*)(&wtile[r * KP + c]) =
            *(const uint4*)(W + (size_t)(n0 + r) * K + c);
    }
    __syncthreads();

    v8f acc[4] = {};
    const float*  Xf = (const float*)Xp;
    const __bf16* Xb = (const __bf16*)Xp;

    #pragma unroll
    for (int kk = 0; kk < K; kk += 32) {
        v16bf a;
        if (XBF) {
            a = frag_bf16(Xb + (size_t)row * K, kk, hi8);
        } else if (ADDX2) {
            a = frag_f32_add(Xf + (size_t)row * K, X2 + (size_t)row * K, kk, hi8);
        } else {
            a = frag_f32(Xf + (size_t)row * K, kk, hi8);
        }

        #pragma unroll
        for (int t = 0; t < 4; ++t) {
            const __bf16* wrow = &wtile[(t * 16 + l15) * KP];
            const v8bf c0 = *(const v8bf*)(wrow + kk + hi8);
            const v8bf c1 = *(const v8bf*)(wrow + kk + 16 + hi8);
            v16bf b;
            #pragma unroll
            for (int j = 0; j < 8; ++j) { b[j] = c0[j]; b[8 + j] = c1[j]; }
            acc[t] = __builtin_amdgcn_wmma_f32_16x16x32_bf16(
                false, a, false, b, (short)0, acc[t], false, false);
        }
    }

    const int rbase = mt * 16 + (hi8 ? 8 : 0);
    #pragma unroll
    for (int t = 0; t < 4; ++t) {
        const int col = n0 + t * 16 + l15;
        const float bv = Bias[col];
        #pragma unroll
        for (int r = 0; r < 8; ++r) {
            float v = acc[t][r] + bv;
            if (relu) v = fmaxf(v, 0.0f);
            if (OUTBF) ((__bf16*)Yp)[(size_t)(rbase + r) * Nout + col] = (__bf16)v;
            else       ((float*)Yp) [(size_t)(rbase + r) * Nout + col] = v;
        }
    }
}

// ---------------------------------------------------------------------------
// f32 -> bf16 row-major conversion (weights, once per launch)
// ---------------------------------------------------------------------------
__global__ __launch_bounds__(256)
void dsvt_cvt_bf16(const float* __restrict__ src, __bf16* __restrict__ dst, int n)
{
    const int i = blockIdx.x * blockDim.x + threadIdx.x;
    if (i < n) dst[i] = (__bf16)src[i];
}

// ---------------------------------------------------------------------------
// Wave32 reduction helper
// ---------------------------------------------------------------------------
__device__ __forceinline__ float wave_sum32(float v)
{
    #pragma unroll
    for (int off = 16; off >= 1; off >>= 1)
        v += __shfl_xor(v, off, 32);
    return v;
}

// ---------------------------------------------------------------------------
// Attention: one block per set (288 thr = 9 waves = (head,query) pairs).
// QK layout: row stride 2C, [Q | K]. Output scattered to voxel order, bf16.
// ---------------------------------------------------------------------------
__global__ __launch_bounds__(288)
void dsvt_attn(const float* __restrict__ QK, const float* __restrict__ V,
               const int* __restrict__ inds, const unsigned char* __restrict__ mask,
               __bf16* __restrict__ Out)
{
    __shared__ int   idx[kSET];
    __shared__ float ks[kSET][kC];
    __shared__ float vs[kSET][kC];

    const int t = threadIdx.x;
    const int s = blockIdx.x;

    if (t < kSET) idx[t] = inds[s * kSET + t];
    __syncthreads();

    for (int i = t; i < kSET * kC; i += 288) {
        const int r = i / kC, c = i - r * kC;
        const size_t vr = (size_t)idx[r];
        ks[r][c] = QK[vr * (2 * kC) + kC + c];
        vs[r][c] = V[vr * kC + c];
    }
    __syncthreads();

    const int h = t / kSET;        // 0..7
    const int q = t - h * kSET;    // 0..35
    const size_t qrow = (size_t)idx[q];

    float qreg[kDH];
    #pragma unroll
    for (int d = 0; d < kDH; ++d)
        qreg[d] = QK[qrow * (2 * kC) + h * kDH + d];

    float sc[kSET];
    float mx = -1e30f;
    #pragma unroll 4
    for (int j = 0; j < kSET; ++j) {
        float acc = 0.0f;
        #pragma unroll
        for (int d = 0; d < kDH; ++d)
            acc += qreg[d] * ks[j][h * kDH + d];
        acc *= kSCALE;
        if (mask[s * kSET + j]) acc = -1e9f;
        sc[j] = acc;
        mx = fmaxf(mx, acc);
    }

    float denom = 0.0f;
    #pragma unroll 4
    for (int j = 0; j < kSET; ++j) { sc[j] = __expf(sc[j] - mx); denom += sc[j]; }
    const float rd = 1.0f / denom;

    #pragma unroll
    for (int d = 0; d < kDH; ++d) {
        float acc = 0.0f;
        #pragma unroll 4
        for (int j = 0; j < kSET; ++j)
            acc += sc[j] * vs[j][h * kDH + d];
        Out[qrow * kC + h * kDH + d] = (__bf16)(acc * rd);
    }
}

// ---------------------------------------------------------------------------
// Y = LayerNorm(A + B) * w + b     — one row per wave (192 = 6 x 32 lanes)
// ---------------------------------------------------------------------------
__global__ __launch_bounds__(256)
void dsvt_add_ln(const float* __restrict__ A, const float* __restrict__ B,
                 const float* __restrict__ w, const float* __restrict__ b,
                 float* __restrict__ Y)
{
    const int lane = threadIdx.x & 31;
    const int row  = blockIdx.x * (blockDim.x >> 5) + (threadIdx.x >> 5);
    const float* ar = A + (size_t)row * kC;
    const float* br = B + (size_t)row * kC;

    float x[6];
    float s = 0.0f;
    #pragma unroll
    for (int i = 0; i < 6; ++i) { x[i] = ar[lane + 32 * i] + br[lane + 32 * i]; s += x[i]; }
    const float m = wave_sum32(s) * (1.0f / kC);
    float v = 0.0f;
    #pragma unroll
    for (int i = 0; i < 6; ++i) { const float d = x[i] - m; v += d * d; }
    const float inv = rsqrtf(wave_sum32(v) * (1.0f / kC) + kEPS);
    #pragma unroll
    for (int i = 0; i < 6; ++i) {
        const int c = lane + 32 * i;
        Y[(size_t)row * kC + c] = (x[i] - m) * inv * w[c] + b[c];
    }
}

// ---------------------------------------------------------------------------
// t = LN(X + F, n2) ; Y = LN(t + Ident, en)   — fused double layernorm
// ---------------------------------------------------------------------------
__global__ __launch_bounds__(256)
void dsvt_ffn_ln2(const float* __restrict__ X, const float* __restrict__ Fv,
                  const float* __restrict__ Iv,
                  const float* __restrict__ w2, const float* __restrict__ b2,
                  const float* __restrict__ we, const float* __restrict__ be,
                  float* __restrict__ Y)
{
    const int lane = threadIdx.x & 31;
    const int row  = blockIdx.x * (blockDim.x >> 5) + (threadIdx.x >> 5);
    const float* xr = X  + (size_t)row * kC;
    const float* fr = Fv + (size_t)row * kC;
    const float* ir = Iv + (size_t)row * kC;

    float x[6];
    float s = 0.0f;
    #pragma unroll
    for (int i = 0; i < 6; ++i) { x[i] = xr[lane + 32 * i] + fr[lane + 32 * i]; s += x[i]; }
    const float m1 = wave_sum32(s) * (1.0f / kC);
    float v1 = 0.0f;
    #pragma unroll
    for (int i = 0; i < 6; ++i) { const float d = x[i] - m1; v1 += d * d; }
    const float inv1 = rsqrtf(wave_sum32(v1) * (1.0f / kC) + kEPS);

    float tt[6];
    float s2 = 0.0f;
    #pragma unroll
    for (int i = 0; i < 6; ++i) {
        const int c = lane + 32 * i;
        tt[i] = (x[i] - m1) * inv1 * w2[c] + b2[c] + ir[c];
        s2 += tt[i];
    }
    const float m2 = wave_sum32(s2) * (1.0f / kC);
    float v2 = 0.0f;
    #pragma unroll
    for (int i = 0; i < 6; ++i) { const float d = tt[i] - m2; v2 += d * d; }
    const float inv2 = rsqrtf(wave_sum32(v2) * (1.0f / kC) + kEPS);
    #pragma unroll
    for (int i = 0; i < 6; ++i) {
        const int c = lane + 32 * i;
        Y[(size_t)row * kC + c] = (tt[i] - m2) * inv2 * we[c] + be[c];
    }
}

// ---------------------------------------------------------------------------
// Host launcher
// ---------------------------------------------------------------------------
extern "C" void kernel_launch(void* const* d_in, const int* in_sizes, int n_in,
                              void* d_out, int out_size, void* d_ws, size_t ws_size,
                              hipStream_t stream)
{
    (void)in_sizes; (void)n_in; (void)out_size; (void)ws_size;

    const float*         pillar = (const float*)d_in[0];
    const int*           inds0  = (const int*)d_in[1];
    const int*           inds1  = (const int*)d_in[2];
    const unsigned char* mask0  = (const unsigned char*)d_in[3];
    const unsigned char* mask1  = (const unsigned char*)d_in[4];
    const float*         pos    = (const float*)d_in[5];
    const float* in_w = (const float*)d_in[6];
    const float* in_b = (const float*)d_in[7];
    const float* ow   = (const float*)d_in[8];
    const float* ob   = (const float*)d_in[9];
    const float* l1w  = (const float*)d_in[10];
    const float* l1b  = (const float*)d_in[11];
    const float* l2w  = (const float*)d_in[12];
    const float* l2b  = (const float*)d_in[13];
    const float* n1w  = (const float*)d_in[14];
    const float* n1b  = (const float*)d_in[15];
    const float* n2w  = (const float*)d_in[16];
    const float* n2b  = (const float*)d_in[17];
    const float* enw  = (const float*)d_in[18];
    const float* enb  = (const float*)d_in[19];
    const float* bnw  = (const float*)d_in[20];
    const float* bnb  = (const float*)d_in[21];
    float* outp = (float*)d_out;

    // ---- workspace layout -------------------------------------------------
    // f32 region, units of NC = N*C floats:
    //   [0]=T0 [1]=T1 [2]=R [3..4]=QK (P reuses [3], X reuses [4])
    //   [5]=V (F2 reuses)  [6]=attnOut(bf16)  [7]=H1(bf16, N*FF)
    // then bf16 weights for all 8 layers (~4.7 MB).
    float* ws = (float*)d_ws;
    const size_t NC = (size_t)kN * kC;
    float*  T0  = ws;
    float*  T1  = ws + NC;
    float*  R   = ws + 2 * NC;
    float*  QKb = ws + 3 * NC;              // 2 units
    float*  Vb  = ws + 5 * NC;
    __bf16* Ab  = (__bf16*)(ws + 6 * NC);   // N*C bf16
    __bf16* Hb  = (__bf16*)(ws + 7 * NC);   // N*FF bf16
    float*  Pb  = QKb;                      // after attention, QK is dead
    float*  Xb  = QKb + NC;                 // after out-proj, K half is dead
    float*  Fb  = Vb;                       // after attention, V is dead

    __bf16* wbf     = (__bf16*)(ws + 8 * NC);
    __bf16* in_w_bf = wbf;                                   // NL*3C*C
    __bf16* ow_bf   = in_w_bf + (size_t)kNL * 3 * kC * kC;   // NL*C*C
    __bf16* l1w_bf  = ow_bf   + (size_t)kNL * kC * kC;       // NL*FF*C
    __bf16* l2w_bf  = l1w_bf  + (size_t)kNL * kFF * kC;      // NL*C*FF

    // ---- weight conversion (runs every call; deterministic) ---------------
    {
        const int n0 = kNL * 3 * kC * kC;   // 884736
        const int n1 = kNL * kC * kC;       // 294912
        const int n2 = kNL * kFF * kC;      // 589824
        dsvt_cvt_bf16<<<n0 / 256, 256, 0, stream>>>(in_w, in_w_bf, n0);
        dsvt_cvt_bf16<<<n1 / 256, 256, 0, stream>>>(ow,   ow_bf,   n1);
        dsvt_cvt_bf16<<<n2 / 256, 256, 0, stream>>>(l1w,  l1w_bf,  n2);
        dsvt_cvt_bf16<<<n2 / 256, 256, 0, stream>>>(l2w,  l2w_bf,  n2);
    }

    const float* cur = pillar;
    const float* res = pillar;
    const int lnGrid = kN / 8;              // 9216 blocks, wave-per-row
    const dim3 gNC (kN / 128, kC  / 64);    // (576, 3)
    const dim3 gNFF(kN / 128, kFF / 64);    // (576, 6)

    for (int blk = 0; blk < 4; ++blk) {
        const int*           indsT = (blk % 2 == 0) ? inds0 : inds1;
        const unsigned char* maskT = (blk % 2 == 0) ? mask0 : mask1;
        for (int s = 0; s < 2; ++s) {
            const int l = 2 * blk + s;
            const float*         posl  = pos   + (size_t)l * NC;
            const int*           indsl = indsT + (size_t)s * kNSET * kSET;
            const unsigned char* maskl = maskT + (size_t)s * kNSET * kSET;

            // Q,K projection: input = cur + pos (fused), weight rows [0 : 2C)
            dsvt_gemm_wmma<false, false, true, kC><<<gNFF, 256, 0, stream>>>(
                cur, posl,
                in_w_bf + (size_t)l * 3 * kC * kC,
                in_b    + (size_t)l * 3 * kC,
                QKb, 2 * kC, 0);

            // V projection: input = cur, weight rows [2C : 3C)
            dsvt_gemm_wmma<false, false, false, kC><<<gNC, 256, 0, stream>>>(
                cur, nullptr,
                in_w_bf + (size_t)l * 3 * kC * kC + (size_t)2 * kC * kC,
                in_b    + (size_t)l * 3 * kC + 2 * kC,
                Vb, kC, 0);

            // Set attention (gather via inds, scatter back, bf16 out)
            dsvt_attn<<<kNSET, 288, 0, stream>>>(QKb, Vb, indsl, maskl, Ab);

            // Output projection (bf16 in, f32 out)
            dsvt_gemm_wmma<true, false, false, kC><<<gNC, 256, 0, stream>>>(
                Ab, nullptr,
                ow_bf + (size_t)l * kC * kC,
                ob    + (size_t)l * kC,
                Pb, kC, 0);

            // x = LN(cur + proj, norm1)
            dsvt_add_ln<<<lnGrid, 256, 0, stream>>>(
                cur, Pb, n1w + l * kC, n1b + l * kC, Xb);

            // h = relu(x @ W1^T + b1)  -> bf16
            dsvt_gemm_wmma<false, true, false, kC><<<gNFF, 256, 0, stream>>>(
                Xb, nullptr,
                l1w_bf + (size_t)l * kFF * kC,
                l1b    + (size_t)l * kFF,
                Hb, kFF, 1);

            // f = h @ W2^T + b2  (bf16 in, f32 out, K = FF)
            dsvt_gemm_wmma<true, false, false, kFF><<<gNC, 256, 0, stream>>>(
                Hb, nullptr,
                l2w_bf + (size_t)l * kC * kFF,
                l2b    + (size_t)l * kC,
                Fb, kC, 0);

            // out = LN( LN(x + f, norm2) + ident, enc_norm )
            float* nxt = (cur == T0) ? T1 : T0;
            dsvt_ffn_ln2<<<lnGrid, 256, 0, stream>>>(
                Xb, Fb, cur,
                n2w + l * kC, n2b + l * kC,
                enw + l * kC, enb + l * kC,
                nxt);
            cur = nxt;
        }
        // block norm: out = LN(res + out, blk_norm); last block -> d_out
        float* tgt = (blk == 3) ? outp : R;
        dsvt_add_ln<<<lnGrid, 256, 0, stream>>>(
            res, cur, bnw + blk * kC, bnb + blk * kC, tgt);
        cur = tgt;
        res = tgt;
    }
}
